// LinearAttentionRNN_43825846288472
// MI455X (gfx1250) — matrix operand order repaired and checked by reading, not compiled, not run on hardware
//
#include <hip/hip_runtime.h>
#include <math.h>

// ---------------------------------------------------------------------------
// Problem constants (reference): B=4, S=4096, D=2048, H=16, DH=128, F=64
// ---------------------------------------------------------------------------
constexpr int Bc  = 4;
constexpr int Sc  = 4096;
constexpr int Dc  = 2048;
constexpr int Hc  = 16;
constexpr int DHc = 128;
constexpr int Fc  = 64;
constexpr int Mc  = Bc * Sc;   // 16384

typedef _Float16 f16;
typedef __attribute__((ext_vector_type(16))) _Float16 v16h;
typedef __attribute__((ext_vector_type(8)))  _Float16 v8h;
typedef __attribute__((ext_vector_type(8)))  float    v8f;

__device__ __forceinline__ v8f zero8() {
  v8f z;
  #pragma unroll
  for (int i = 0; i < 8; ++i) z[i] = 0.0f;
  return z;
}

// D = A(16x32 f16) * B(32x16 f16) + C(16x16 f32)
__device__ __forceinline__ v8f wmma32(v16h a, v16h b, v8f c) {
  return __builtin_amdgcn_wmma_f32_16x16x32_f16(false, a, false, b,
                                                (short)0, c, false, false);
}

// ---------------------------------------------------------------------------
// Fragment loaders (wave32 layouts per CDNA5 ISA §7.12.2).
// All strides are multiples of 8 halfs so every v8h access is 16B aligned.
// ---------------------------------------------------------------------------
__device__ __forceinline__ v16h ldsA(const f16* base, int stride, int mBase, int kOff) {
  int lane = threadIdx.x & 31;
  int m  = mBase + (lane & 15);
  int kb = (lane >> 4) * 8;
  const f16* p = base + (size_t)m * stride + kOff + kb;
  v8h lo = *(const v8h*)(p);        // K = kOff+kb .. +7
  v8h hi = *(const v8h*)(p + 16);   // K = kOff+16+kb .. +7
  v16h r;
  #pragma unroll
  for (int i = 0; i < 8; ++i) { r[i] = lo[i]; r[i + 8] = hi[i]; }
  return r;
}

__device__ __forceinline__ v16h ldsB(const f16* base, int stride, int nBase, int kOff) {
  int lane = threadIdx.x & 31;
  int n  = nBase + (lane & 15);
  int kb = (lane >> 4) * 8;
  const f16* p = base + (size_t)n * stride + kOff + kb;
  v8h lo = *(const v8h*)(p);
  v8h hi = *(const v8h*)(p + 16);
  v16h r;
  #pragma unroll
  for (int i = 0; i < 8; ++i) { r[i] = lo[i]; r[i + 8] = hi[i]; }
  return r;
}

// ---------------------------------------------------------------------------
// One-time precision conversion kernels (amortized across 4 big GEMMs).
// ---------------------------------------------------------------------------
__global__ __launch_bounds__(256) void convert_kernel(
    const float* __restrict__ src, f16* __restrict__ dst) {
  size_t i = ((size_t)blockIdx.x * 256 + threadIdx.x) * 8;
  float4 a = *(const float4*)(src + i);
  float4 b = *(const float4*)(src + i + 4);
  v8h o;
  o[0] = (f16)a.x; o[1] = (f16)a.y; o[2] = (f16)a.z; o[3] = (f16)a.w;
  o[4] = (f16)b.x; o[5] = (f16)b.y; o[6] = (f16)b.z; o[7] = (f16)b.w;
  *(v8h*)(dst + i) = o;
}

// W (D,D) f32 row-major -> Wt (D,D) f16 with Wt[n][k] = W[k][n]
__global__ __launch_bounds__(256) void transpose_convert_kernel(
    const float* __restrict__ W, f16* __restrict__ Wt) {
  __shared__ float tile[32][33];
  const int k0 = blockIdx.x * 32, n0 = blockIdx.y * 32;
  const int tr = threadIdx.x >> 5;   // 0..7
  const int tc = threadIdx.x & 31;
  #pragma unroll
  for (int i = 0; i < 4; ++i)
    tile[tr + i * 8][tc] = W[(size_t)(k0 + tr + i * 8) * Dc + n0 + tc];
  __syncthreads();
  #pragma unroll
  for (int i = 0; i < 4; ++i)
    Wt[(size_t)(n0 + tr + i * 8) * Dc + k0 + tc] = (f16)tile[tc][tr + i * 8];
}

// ---------------------------------------------------------------------------
// Kernel 1: C[M,N] = A[M,K] @ Wt[N,K]^T + bias[N]
// A, Wt are f16; accumulate f32 in WMMA; OutT selects f16 (intermediates) or
// f32 (final output). Double-buffered LDS, registers stage the next tile so
// global-load latency overlaps the 8 WMMAs per 32-K step.
// ---------------------------------------------------------------------------
#define BM 128
#define BN 128
#define BK 32
#define SA 40   // LDS k-stride (halfs), padded
#define SB 40

template <typename OutT>
__global__ __launch_bounds__(256) void gemm_f16_kernel(
    const f16* __restrict__ A,    // (M,K)
    const f16* __restrict__ Wt,   // (N,K)  (pre-transposed weight)
    const float* __restrict__ bias,
    OutT* __restrict__ C,         // (M,N)
    int M, int N, int K) {
  __shared__ f16 lA[2][BM * SA];
  __shared__ f16 lB[2][BN * SB];

  const int bm = blockIdx.x * BM;
  const int bn = blockIdx.y * BN;
  const int tid  = threadIdx.x;
  const int wave = tid >> 5;
  const int lane = tid & 31;
  const int wm = (wave & 3) * 32;
  const int wn = (wave >> 2) * 64;

  // Staging: 512 v8h chunks per 128x32 tile, 2 per thread.
  int ar[2], ac[2];
  #pragma unroll
  for (int i = 0; i < 2; ++i) {
    int idx = tid + i * 256;
    ar[i] = idx >> 2;          // row (m or n), 4 chunks of 8 halfs per row
    ac[i] = (idx & 3) * 8;     // k offset
  }

  v8f acc[2][4];
  #pragma unroll
  for (int r = 0; r < 2; ++r)
    #pragma unroll
    for (int c = 0; c < 4; ++c) acc[r][c] = zero8();

  v8h ra[2], rb[2];
  // load tile (k0) into registers
  auto gload = [&](int k0) {
    #pragma unroll
    for (int i = 0; i < 2; ++i) {
      ra[i] = *(const v8h*)(A  + (size_t)(bm + ar[i]) * K + k0 + ac[i]);
      rb[i] = *(const v8h*)(Wt + (size_t)(bn + ar[i]) * K + k0 + ac[i]);
    }
  };
  auto lstore = [&](int buf) {
    #pragma unroll
    for (int i = 0; i < 2; ++i) {
      *(v8h*)(&lA[buf][(size_t)ar[i] * SA + ac[i]]) = ra[i];
      *(v8h*)(&lB[buf][(size_t)ar[i] * SB + ac[i]]) = rb[i];
    }
  };

  gload(0);
  int cur = 0;
  for (int k0 = 0; k0 < K; k0 += BK) {
    lstore(cur);
    __syncthreads();
    if (k0 + BK < K) gload(k0 + BK);
    if (k0 + 2 * BK < K) {   // gfx1250 global_prefetch_b8 for tile after next
      __builtin_prefetch(A  + (size_t)(bm + ar[0]) * K + k0 + 2 * BK, 0, 1);
      __builtin_prefetch(Wt + (size_t)(bn + ar[0]) * K + k0 + 2 * BK, 0, 1);
    }

    v16h af[2], bf[4];
    #pragma unroll
    for (int r = 0; r < 2; ++r) af[r] = ldsA(lA[cur], SA, wm + r * 16, 0);
    #pragma unroll
    for (int c = 0; c < 4; ++c) bf[c] = ldsB(lB[cur], SB, wn + c * 16, 0);
    #pragma unroll
    for (int r = 0; r < 2; ++r)
      #pragma unroll
      for (int c = 0; c < 4; ++c)
        acc[r][c] = wmma32(af[r], bf[c], acc[r][c]);

    cur ^= 1;
  }

  const int colL  = lane & 15;
  const int rowHi = (lane >> 4) * 8;
  #pragma unroll
  for (int r = 0; r < 2; ++r) {
    #pragma unroll
    for (int c = 0; c < 4; ++c) {
      int n = bn + wn + c * 16 + colL;
      int m = bm + wm + r * 16 + rowHi;
      float bb = bias[n];
      #pragma unroll
      for (int v = 0; v < 8; ++v)
        C[(size_t)(m + v) * N + n] = (OutT)(acc[r][c][v] + bb);
    }
  }
}

// ---------------------------------------------------------------------------
// Kernel 2: feature map per (b,h): Out = elu(X @ fw1 + fb1) @ fw2 + fb2
// X = P[b, s0:s0+64, h*DH:h*DH+DH] (f16), Out (BH,S,F) f16.
// ---------------------------------------------------------------------------
#define SX 136
#define ST 72

__global__ __launch_bounds__(128) void feature_map_kernel(
    const f16* __restrict__ P,       // (B,S,D) f16
    const float* __restrict__ fw1,   // (DH,F) f32
    const float* __restrict__ fb1,
    const float* __restrict__ fw2,   // (F,F) f32
    const float* __restrict__ fb2,
    f16* __restrict__ Out) {         // (B*H, S, F) f16
  __shared__ f16 lX [64 * SX];   // [s][dh]
  __shared__ f16 w1t[Fc * SX];   // [f][dh]
  __shared__ f16 t1 [64 * ST];   // [s][f1]
  __shared__ f16 w2t[Fc * ST];   // [f2][f1]

  const int s0 = blockIdx.x * 64;
  const int bh = blockIdx.y;
  const int b = bh / Hc, h = bh % Hc;
  const int tid = threadIdx.x;

  // X tile: 64x128 halfs = 1024 v8h chunks, 8 per thread (pure copy)
  #pragma unroll
  for (int i = 0; i < 8; ++i) {
    int idx = tid + i * 128;
    int r  = idx >> 4;            // 16 chunks per row
    int c8 = (idx & 15) * 8;
    *(v8h*)(&lX[(size_t)r * SX + c8]) =
        *(const v8h*)(P + ((size_t)b * Sc + s0 + r) * Dc + h * DHc + c8);
  }
  // fw1 (128x64 f32) -> w1t[f][dh]
  #pragma unroll
  for (int i = 0; i < 16; ++i) {
    int idx = tid + i * 128;
    int dh = idx >> 4;
    int c4 = (idx & 15) * 4;
    float4 v = *(const float4*)(fw1 + (size_t)dh * Fc + c4);
    w1t[(size_t)(c4 + 0) * SX + dh] = (f16)v.x;
    w1t[(size_t)(c4 + 1) * SX + dh] = (f16)v.y;
    w1t[(size_t)(c4 + 2) * SX + dh] = (f16)v.z;
    w1t[(size_t)(c4 + 3) * SX + dh] = (f16)v.w;
  }
  // fw2 (64x64 f32) -> w2t[f2][f1]
  #pragma unroll
  for (int i = 0; i < 8; ++i) {
    int idx = tid + i * 128;
    int f1 = idx >> 4;
    int c4 = (idx & 15) * 4;
    float4 v = *(const float4*)(fw2 + (size_t)f1 * Fc + c4);
    w2t[(size_t)(c4 + 0) * ST + f1] = (f16)v.x;
    w2t[(size_t)(c4 + 1) * ST + f1] = (f16)v.y;
    w2t[(size_t)(c4 + 2) * ST + f1] = (f16)v.z;
    w2t[(size_t)(c4 + 3) * ST + f1] = (f16)v.w;
  }
  __syncthreads();

  const int wave  = tid >> 5;
  const int lane  = tid & 31;
  const int wm    = wave * 16;
  const int colL  = lane & 15;
  const int rowHi = (lane >> 4) * 8;

  // Stage 1: T1 = elu(X @ fw1 + fb1)
  #pragma unroll
  for (int c = 0; c < 4; ++c) {
    v8f acc = zero8();
    #pragma unroll
    for (int kb = 0; kb < DHc; kb += 32)
      acc = wmma32(ldsA(lX, SX, wm, kb), ldsB(w1t, SX, c * 16, kb), acc);
    int f1 = c * 16 + colL;
    float bb = fb1[f1];
    #pragma unroll
    for (int v = 0; v < 8; ++v) {
      float x = acc[v] + bb;
      x = (x > 0.0f) ? x : (__expf(x) - 1.0f);   // ELU (alpha = 1)
      t1[(size_t)(wm + rowHi + v) * ST + f1] = (f16)x;
    }
  }
  __syncthreads();

  // Stage 2: Out = T1 @ fw2 + fb2
  #pragma unroll
  for (int c = 0; c < 4; ++c) {
    v8f acc = zero8();
    #pragma unroll
    for (int kb = 0; kb < Fc; kb += 32)
      acc = wmma32(ldsA(t1, ST, wm, kb), ldsB(w2t, ST, c * 16, kb), acc);
    int f2 = c * 16 + colL;
    float bb = fb2[f2];
    #pragma unroll
    for (int v = 0; v < 8; ++v)
      Out[((size_t)bh * Sc + s0 + wm + rowHi + v) * Fc + f2] = (f16)(acc[v] + bb);
  }
}

// ---------------------------------------------------------------------------
// Kernel 3: KV[bh] = Km[bh]^T (F x S) @ V[bh] (S x DH) -> (64,128) f32.
// One block per (b,h), loops S in chunks of 32.
// ---------------------------------------------------------------------------
__global__ __launch_bounds__(256) void kv_kernel(
    const f16* __restrict__ Km,   // (BH,S,F) f16
    const f16* __restrict__ V,    // (B,S,D)  f16
    float* __restrict__ KV) {     // (BH,F,DH) f32
  __shared__ f16 lK[Fc  * 40];   // [f][s]
  __shared__ f16 lV[DHc * 40];   // [d][s]

  const int bh = blockIdx.x;
  const int b = bh / Hc, h = bh % Hc;
  const int tid  = threadIdx.x;
  const int wave = tid >> 5;
  const int lane = tid & 31;
  const int wf = (wave & 3) * 16;
  const int wd = (wave >> 2) * 64;

  v8f acc[4];
  #pragma unroll
  for (int c = 0; c < 4; ++c) acc[c] = zero8();

  for (int s0 = 0; s0 < Sc; s0 += 32) {
    // Km chunk 32x64 halfs = 256 v8h, 1 per thread; transpose to [f][s]
    {
      int sr = tid >> 3;
      int c8 = (tid & 7) * 8;
      v8h v = *(const v8h*)(Km + ((size_t)bh * Sc + s0 + sr) * Fc + c8);
      #pragma unroll
      for (int j = 0; j < 8; ++j) lK[(size_t)(c8 + j) * 40 + sr] = v[j];
    }
    // V chunk 32x128 halfs = 512 v8h, 2 per thread; transpose to [d][s]
    #pragma unroll
    for (int i = 0; i < 2; ++i) {
      int idx = tid + i * 256;
      int sr = idx >> 4;
      int c8 = (idx & 15) * 8;
      v8h v = *(const v8h*)(V + ((size_t)b * Sc + s0 + sr) * Dc + h * DHc + c8);
      #pragma unroll
      for (int j = 0; j < 8; ++j) lV[(size_t)(c8 + j) * 40 + sr] = v[j];
    }
    __syncthreads();

    v16h af = ldsA(lK, 40, wf, 0);    // A = Km^T: rows f, k = s
    #pragma unroll
    for (int c = 0; c < 4; ++c)
      acc[c] = wmma32(af, ldsB(lV, 40, wd + c * 16, 0), acc[c]);

    __syncthreads();
  }

  const int colL  = lane & 15;
  const int rowHi = (lane >> 4) * 8;
  #pragma unroll
  for (int c = 0; c < 4; ++c) {
    int d = wd + c * 16 + colL;
    int f = wf + rowHi;
    #pragma unroll
    for (int v = 0; v < 8; ++v)
      KV[(size_t)bh * Fc * DHc + (size_t)(f + v) * DHc + d] = acc[c][v];
  }
}

// Z[bh,f] = sum_s Km[bh,s,f]  (f32 accumulate)
__global__ __launch_bounds__(64) void z_kernel(
    const f16* __restrict__ Km, float* __restrict__ Z) {
  int bh = blockIdx.x;
  int f  = threadIdx.x;
  const f16* p = Km + (size_t)bh * Sc * Fc + f;
  float s = 0.0f;
  for (int i = 0; i < Sc; ++i) s += (float)p[(size_t)i * Fc];
  Z[bh * Fc + f] = s;
}

// ---------------------------------------------------------------------------
// Kernel 4: attn = (Qm @ KV) / max(Qm @ Z, 1e-6), written f16 in (B,S,D).
// ---------------------------------------------------------------------------
__global__ __launch_bounds__(256) void attn_kernel(
    const f16* __restrict__ Qm,    // (BH,S,F) f16
    const float* __restrict__ KV,  // (BH,F,DH) f32
    const float* __restrict__ Z,   // (BH,F) f32
    f16* __restrict__ Out) {       // (B,S,D) f16
  __shared__ f16   lQ [64  * ST];  // [s][f]
  __shared__ f16   lKV[DHc * ST];  // [d][f]
  __shared__ float lZ[Fc];
  __shared__ float lden[64];

  const int s0 = blockIdx.x * 64;
  const int bh = blockIdx.y;
  const int b = bh / Hc, h = bh % Hc;
  const int tid = threadIdx.x;

  // Qm tile 64x64 halfs = 512 v8h, 2 per thread (pure copy, A layout)
  #pragma unroll
  for (int i = 0; i < 2; ++i) {
    int idx = tid + i * 256;
    int r  = idx >> 3;            // 8 chunks per row
    int c8 = (idx & 7) * 8;
    *(v8h*)(&lQ[(size_t)r * ST + c8]) =
        *(const v8h*)(Qm + ((size_t)bh * Sc + s0 + r) * Fc + c8);
  }
  // KV (64x128 f32) -> lKV[d][f]: 2048 float4, 8 per thread
  #pragma unroll
  for (int i = 0; i < 8; ++i) {
    int idx = tid + i * 256;
    int fr = idx >> 5;
    int c4 = (idx & 31) * 4;
    float4 v = *(const float4*)(KV + (size_t)bh * Fc * DHc + (size_t)fr * DHc + c4);
    lKV[(size_t)(c4 + 0) * ST + fr] = (f16)v.x;
    lKV[(size_t)(c4 + 1) * ST + fr] = (f16)v.y;
    lKV[(size_t)(c4 + 2) * ST + fr] = (f16)v.z;
    lKV[(size_t)(c4 + 3) * ST + fr] = (f16)v.w;
  }
  if (tid < Fc) lZ[tid] = Z[bh * Fc + tid];
  __syncthreads();

  if (tid < 64) {
    float sden = 0.0f;
    #pragma unroll
    for (int f = 0; f < Fc; ++f)
      sden += (float)lQ[(size_t)tid * ST + f] * lZ[f];
    lden[tid] = fmaxf(sden, 1e-6f);
  }
  __syncthreads();

  const int wave  = tid >> 5;
  const int lane  = tid & 31;
  const int wm = (wave & 3) * 16;
  const int wd = (wave >> 2) * 64;
  const int colL  = lane & 15;
  const int rowHi = (lane >> 4) * 8;

  #pragma unroll
  for (int c = 0; c < 4; ++c) {
    v8f acc = zero8();
    #pragma unroll
    for (int kb = 0; kb < Fc; kb += 32)
      acc = wmma32(ldsA(lQ, ST, wm, kb), ldsB(lKV, ST, wd + c * 16, kb), acc);
    int d = wd + c * 16 + colL;
    #pragma unroll
    for (int v = 0; v < 8; ++v) {
      int r = wm + rowHi + v;
      Out[((size_t)b * Sc + s0 + r) * Dc + h * DHc + d] = (f16)(acc[v] / lden[r]);
    }
  }
}

// ---------------------------------------------------------------------------
// Launch. Inputs: x, qw, qb, kw, kb, vw, vb, fw1, fb1, fw2, fb2, ow, ob (f32).
// Workspace (f16 region first, then f32):
//   xh   : M*D          f16
//   Qh   : M*D          f16  (reused as attn buffer)
//   Kh   : M*D          f16
//   Vh   : M*D          f16
//   qwT,kwT,vwT,owT : D*D f16 each (weights transposed to [N,K])
//   Qmh  : BH*S*F       f16
//   Kmh  : BH*S*F       f16
//   KV   : BH*F*DH      f32
//   Zb   : BH*F         f32
// Total ~371 MB scratch.
// ---------------------------------------------------------------------------
extern "C" void kernel_launch(void* const* d_in, const int* in_sizes, int n_in,
                              void* d_out, int out_size, void* d_ws, size_t ws_size,
                              hipStream_t stream) {
  (void)in_sizes; (void)n_in; (void)out_size; (void)ws_size;
  const float* x     = (const float*)d_in[0];
  const float* qw    = (const float*)d_in[1];
  const float* qb    = (const float*)d_in[2];
  const float* kw    = (const float*)d_in[3];
  const float* kbias = (const float*)d_in[4];
  const float* vw    = (const float*)d_in[5];
  const float* vb    = (const float*)d_in[6];
  const float* fw1   = (const float*)d_in[7];
  const float* fb1   = (const float*)d_in[8];
  const float* fw2   = (const float*)d_in[9];
  const float* fb2   = (const float*)d_in[10];
  const float* ow    = (const float*)d_in[11];
  const float* ob    = (const float*)d_in[12];
  float* out = (float*)d_out;

  const size_t MD  = (size_t)Mc * Dc;          // 33,554,432
  const size_t DD  = (size_t)Dc * Dc;          //  4,194,304
  const size_t QMS = (size_t)Bc * Hc * Sc * Fc;// 16,777,216

  f16* ws16 = (f16*)d_ws;
  f16* xh   = ws16;
  f16* Qh   = xh  + MD;
  f16* Kh   = Qh  + MD;
  f16* Vh   = Kh  + MD;
  f16* qwT  = Vh  + MD;
  f16* kwT  = qwT + DD;
  f16* vwT  = kwT + DD;
  f16* owT  = vwT + DD;
  f16* Qmh  = owT + DD;
  f16* Kmh  = Qmh + QMS;
  float* KV = (float*)(Kmh + QMS);
  float* Zb = KV + (size_t)Bc * Hc * Fc * DHc;
  f16* attnh = Qh;   // Qh dead after feature_map; reuse for attn

  // One-time f16 conversion + weight transposes
  convert_kernel<<<dim3((unsigned)(MD / 8 / 256)), 256, 0, stream>>>(x, xh);
  dim3 gT(Dc / 32, Dc / 32);
  transpose_convert_kernel<<<gT, 256, 0, stream>>>(qw, qwT);
  transpose_convert_kernel<<<gT, 256, 0, stream>>>(kw, kwT);
  transpose_convert_kernel<<<gT, 256, 0, stream>>>(vw, vwT);
  transpose_convert_kernel<<<gT, 256, 0, stream>>>(ow, owT);

  dim3 gGemm(Mc / BM, Dc / BN);     // (128, 16)
  dim3 gFmap(Sc / 64, Bc * Hc);     // (64, 64)

  // Projections (f16 out)
  gemm_f16_kernel<f16><<<gGemm, 256, 0, stream>>>(xh, qwT, qb,    Qh, Mc, Dc, Dc);
  gemm_f16_kernel<f16><<<gGemm, 256, 0, stream>>>(xh, kwT, kbias, Kh, Mc, Dc, Dc);
  gemm_f16_kernel<f16><<<gGemm, 256, 0, stream>>>(xh, vwT, vb,    Vh, Mc, Dc, Dc);

  // Feature maps (f16 out)
  feature_map_kernel<<<gFmap, 128, 0, stream>>>(Qh, fw1, fb1, fw2, fb2, Qmh);
  feature_map_kernel<<<gFmap, 128, 0, stream>>>(Kh, fw1, fb1, fw2, fb2, Kmh);

  // KV and Z reductions
  kv_kernel<<<dim3(Bc * Hc), 256, 0, stream>>>(Kmh, Vh, KV);
  z_kernel <<<dim3(Bc * Hc),  64, 0, stream>>>(Kmh, Zb);

  // attn (f16, (B,S,D) layout in reused Qh)
  attn_kernel<<<gFmap, 256, 0, stream>>>(Qmh, KV, Zb, attnh);

  // Output projection (f32 out)
  gemm_f16_kernel<float><<<gGemm, 256, 0, stream>>>(attnh, owT, ob, out, Mc, Dc, Dc);
}
